// MambaBlock_9809705304252
// MI455X (gfx1250) — matrix operand order repaired and checked by reading, not compiled, not run on hardware
//
#include <hip/hip_runtime.h>
#include <hip/hip_bf16.h>
#include <math.h>

// ---------------- problem constants (match reference) ----------------
#define D_MODEL 1024
#define D_STATE 16
#define D_CONV  4
#define D_INNER 2048          // EXPAND * D_MODEL
#define DT_RANK 64            // D_MODEL / 16
#define B_SZ    2
#define SEQ_L   1024
#define NTOK    (B_SZ * SEQ_L)   // 2048 rows for all GEMMs
#define DBC_N   (DT_RANK + 2 * D_STATE)  // 96

typedef _Float16 half_t;
typedef __attribute__((ext_vector_type(16))) _Float16 v16h;
typedef __attribute__((ext_vector_type(8)))  _Float16 v8h;
typedef __attribute__((ext_vector_type(8)))  float    v8f;
typedef __attribute__((ext_vector_type(4)))  unsigned int v4u;
typedef __attribute__((ext_vector_type(8)))  int          v8i;
typedef __attribute__((ext_vector_type(4)))  int          v4i;

// ---------------- small helpers ----------------
__device__ __forceinline__ float silu_f(float x) {
    return x / (1.0f + expf(-x));
}
__device__ __forceinline__ float softplus_f(float x) {
    return (x > 20.0f) ? x : log1pf(expf(x));
}

// ---------------- TDM: 2-D tile DMA global -> LDS (f16 elements) ----------------
// Descriptor bit layout per CDNA5 ISA 8.3/8.4 (D# group0 128b, group1 256b).
// dims/strides in data_size units; tensor dims = remaining extent from tile start
// so OOB rows/cols return zero. 6-arg builtin form (clang-23 / therock-10.0).
__device__ __forceinline__ void tdm_load_tile_f16(
    unsigned int lds_addr, const half_t* gptr,
    unsigned int tens_w,          // tensor_dim0: valid elements per row from tile start
    unsigned int tens_h,          // tensor_dim1: valid rows from tile start
    unsigned int tile_w,          // tile_dim0 (elements)
    unsigned int tile_h,          // tile_dim1 (rows)
    unsigned long long row_stride) // tensor_dim0_stride (elements)
{
    unsigned long long ga = (unsigned long long)(size_t)gptr;
    v4u g0;
    g0.x = 1u;                                              // count=1 (valid), user mode
    g0.y = lds_addr;                                        // LDS byte address
    g0.z = (unsigned int)(ga & 0xFFFFFFFFu);                // global_addr[31:0]
    g0.w = (unsigned int)((ga >> 32) & 0x1FFFFFFu)          // global_addr[56:32]
         | (2u << 30);                                      // type = 2 ("image")
    v8i g1;
    g1[0] = (int)(1u << 16);                                // data_size=1 -> 2 bytes
    g1[1] = (int)((tens_w & 0xFFFFu) << 16);                // tensor_dim0[15:0]
    g1[2] = (int)((tens_w >> 16) | ((tens_h & 0xFFFFu) << 16));
    g1[3] = (int)((tens_h >> 16) | (tile_w << 16));         // tile_dim0
    g1[4] = (int)(tile_h & 0xFFFFu);                        // tile_dim1 (tile_dim2=0)
    g1[5] = (int)(row_stride & 0xFFFFFFFFull);              // tensor_dim0_stride lo
    g1[6] = (int)((row_stride >> 32) & 0xFFFFull);          // stride hi (dim1 stride unused, 2-D)
    g1[7] = 0;
    v4i z4 = {0, 0, 0, 0};
    v8i z8 = {0, 0, 0, 0, 0, 0, 0, 0};
    __builtin_amdgcn_tensor_load_to_lds(g0, g1, z4, z4, z8, 0);
}

// ---------------- pack: f32 [R][C] -> f16 transposed [C][R] ----------------
__global__ void k_transpose_f32_to_f16(const float* __restrict__ src, half_t* __restrict__ dst,
                                       int R, int C) {
    long long tid = (long long)blockIdx.x * blockDim.x + threadIdx.x;
    long long total = (long long)R * C;
    if (tid >= total) return;
    int c = (int)(tid / R);
    int r = (int)(tid % R);
    dst[tid] = (half_t)src[(long long)r * C + c];
}

// ---------------- LayerNorm: one block per token, write f16 xn ----------------
__global__ __launch_bounds__(256)
void k_layernorm(const float* __restrict__ x, const float* __restrict__ gamma,
                 const float* __restrict__ beta, half_t* __restrict__ xn) {
    __shared__ float sred[256];
    const int row = blockIdx.x;                       // 0 .. NTOK-1
    const float* xr = x + (long long)row * D_MODEL;

    float s = 0.0f;
    for (int i = threadIdx.x; i < D_MODEL; i += 256) s += xr[i];
    sred[threadIdx.x] = s;
    __syncthreads();
    for (int off = 128; off > 0; off >>= 1) {
        if (threadIdx.x < off) sred[threadIdx.x] += sred[threadIdx.x + off];
        __syncthreads();
    }
    const float mu = sred[0] * (1.0f / D_MODEL);
    __syncthreads();

    float v = 0.0f;
    for (int i = threadIdx.x; i < D_MODEL; i += 256) {
        float d = xr[i] - mu;
        v += d * d;
    }
    sred[threadIdx.x] = v;
    __syncthreads();
    for (int off = 128; off > 0; off >>= 1) {
        if (threadIdx.x < off) sred[threadIdx.x] += sred[threadIdx.x + off];
        __syncthreads();
    }
    const float inv = rsqrtf(sred[0] * (1.0f / D_MODEL) + 1e-5f);

    half_t* xo = xn + (long long)row * D_MODEL;
    for (int i = threadIdx.x; i < D_MODEL; i += 256)
        xo[i] = (half_t)((xr[i] - mu) * inv * gamma[i] + beta[i]);
}

// ---------------- WMMA GEMM with TDM-staged LDS tiles ----------------
// C[M,N] = A[M,K] (f16 row-major) @ Bt[N,K] (f16, pre-transposed) [+ resid]
// Block = 8 waves, 64x64 C tile: wave w -> tile row w&3, tile cols (w>>2)*2 + {0,1}.
// Per 32-wide k-step: wave 0 issues two TENSOR_LOAD_TO_LDS (A 64x32, B 64x32),
// waits TENSORcnt, then all waves consume fragments from LDS via ds_load_b128.
__global__ __launch_bounds__(256)
void k_gemm_wmma(const half_t* __restrict__ A, const half_t* __restrict__ Bt,
                 float* __restrict__ C, const float* __restrict__ resid,
                 int M, int N, int K) {
    __shared__ half_t shA[64 * 32];
    __shared__ half_t shB[64 * 32];

    const int lane = threadIdx.x;            // 0..31
    const int wave = threadIdx.y;            // 0..7
    const int wm   = wave & 3;               // 16-row tile within block
    const int wnb  = (wave >> 2) * 2;        // base 16-col tile within block
    const int m0   = blockIdx.x * 64;
    const int n0   = blockIdx.y * 64;
    const int h16  = lane >> 4;              // 0: lanes 0-15, 1: lanes 16-31
    const int r    = lane & 15;

    v8f c0 = {}, c1 = {};

    const unsigned int ldsA = (unsigned int)(size_t)(void*)&shA[0];
    const unsigned int ldsB = (unsigned int)(size_t)(void*)&shB[0];

    for (int k0 = 0; k0 < K; k0 += 32) {
        __syncthreads();                     // previous iter's LDS reads complete
        if (wave == 0) {
            tdm_load_tile_f16(ldsA, A + (long long)m0 * K + k0,
                              (unsigned)(K - k0), (unsigned)(M - m0), 32, 64,
                              (unsigned long long)K);
            tdm_load_tile_f16(ldsB, Bt + (long long)n0 * K + k0,
                              (unsigned)(K - k0), (unsigned)(N - n0), 32, 64,
                              (unsigned long long)K);
            __builtin_amdgcn_s_wait_tensorcnt(0);
        }
        __syncthreads();                     // tiles visible to all waves

        union { v16h v; v8h h[2]; } a, b0, b1;
        // A 16x32 layout: lanes 0-15 -> K 0..15; lanes 16-31 -> K 16..31
        const half_t* arow = &shA[(wm * 16 + r) * 32];
        a.h[0] = *(const v8h*)(arow + h16 * 16);
        a.h[1] = *(const v8h*)(arow + h16 * 16 + 8);
        // B 32x16: lanes 0-15 -> K 0..15; lanes 16-31 -> K 16..31 (rows of Bt)
        const half_t* brow0 = &shB[((wnb + 0) * 16 + r) * 32];
        const half_t* brow1 = &shB[((wnb + 1) * 16 + r) * 32];
        b0.h[0] = *(const v8h*)(brow0 + h16 * 16);
        b0.h[1] = *(const v8h*)(brow0 + h16 * 16 + 8);
        b1.h[0] = *(const v8h*)(brow1 + h16 * 16);
        b1.h[1] = *(const v8h*)(brow1 + h16 * 16 + 8);

        c0 = __builtin_amdgcn_wmma_f32_16x16x32_f16(false, a.v, false, b0.v,
                                                    (short)0, c0, false, false);
        c1 = __builtin_amdgcn_wmma_f32_16x16x32_f16(false, a.v, false, b1.v,
                                                    (short)0, c1, false, false);
    }

    // D 16x16 f32 layout: VGPR i -> M = h16*8 + i, N = lane&15
#pragma unroll
    for (int i = 0; i < 8; ++i) {
        int gm  = m0 + wm * 16 + h16 * 8 + i;
        int gn0 = n0 + (wnb + 0) * 16 + r;
        int gn1 = n0 + (wnb + 1) * 16 + r;
        if (gm < M) {
            if (gn0 < N) {
                long long idx = (long long)gm * N + gn0;
                float v = c0[i];
                if (resid) v += resid[idx];
                C[idx] = v;
            }
            if (gn1 < N) {
                long long idx = (long long)gm * N + gn1;
                float v = c1[i];
                if (resid) v += resid[idx];
                C[idx] = v;
            }
        }
    }
}

// ---------------- depthwise causal conv1d + SiLU ----------------
__global__ void k_conv_silu(const float* __restrict__ xz, const float* __restrict__ conv_w,
                            const float* __restrict__ conv_b,
                            float* __restrict__ xs_f, half_t* __restrict__ xs_h) {
    long long tid = (long long)blockIdx.x * blockDim.x + threadIdx.x;
    if (tid >= (long long)NTOK * D_INNER) return;
    const int d    = (int)(tid % D_INNER);
    const long long row = tid / D_INNER;   // b*L + l
    const int l    = (int)(row % SEQ_L);

    float acc = conv_b[d];
#pragma unroll
    for (int j = 0; j < D_CONV; ++j) {
        int t = l - (D_CONV - 1) + j;
        if (t >= 0)
            acc += conv_w[d * D_CONV + j] *
                   xz[(row - l + t) * (long long)(2 * D_INNER) + d];
    }
    float s = silu_f(acc);
    xs_f[tid] = s;
    xs_h[tid] = (half_t)s;
}

// ---------------- extract dt columns (first 64 of dbc) as f16 ----------------
__global__ void k_extract_dt(const float* __restrict__ dbc, half_t* __restrict__ dt_h) {
    int tid = blockIdx.x * blockDim.x + threadIdx.x;  // over NTOK*DT_RANK
    if (tid >= NTOK * DT_RANK) return;
    int row = tid / DT_RANK;
    int j   = tid % DT_RANK;
    dt_h[tid] = (half_t)dbc[(long long)row * DBC_N + j];
}

// ---------------- delta = softplus(draw + b_dt) ----------------
__global__ void k_delta(const float* __restrict__ draw, const float* __restrict__ b_dt,
                        float* __restrict__ delta) {
    long long tid = (long long)blockIdx.x * blockDim.x + threadIdx.x;
    if (tid >= (long long)NTOK * D_INNER) return;
    int d = (int)(tid % D_INNER);
    delta[tid] = softplus_f(draw[tid] + b_dt[d]);
}

// ---------------- selective scan + D skip + SiLU(z) gate -> f16 for out GEMM ----
__global__ __launch_bounds__(256)
void k_scan(const float* __restrict__ xs_f, const float* __restrict__ delta_f,
            const float* __restrict__ dbc, const float* __restrict__ A_log,
            const float* __restrict__ Dp, const float* __restrict__ xz,
            half_t* __restrict__ y_h) {
    int tid = blockIdx.x * blockDim.x + threadIdx.x;   // 0 .. B*Di-1
    if (tid >= B_SZ * D_INNER) return;
    const int d = tid % D_INNER;
    const int b = tid / D_INNER;

    float Arow[D_STATE];
#pragma unroll
    for (int n = 0; n < D_STATE; ++n)
        Arow[n] = -expf(A_log[(long long)d * D_STATE + n]);
    const float Dd = Dp[d];

    float h[D_STATE];
#pragma unroll
    for (int n = 0; n < D_STATE; ++n) h[n] = 0.0f;

    for (int l = 0; l < SEQ_L; ++l) {
        const long long row = (long long)b * SEQ_L + l;
        const float dl = delta_f[row * D_INNER + d];
        const float u  = xs_f[row * D_INNER + d];
        const float du = dl * u;
        const float* bc = dbc + row * DBC_N;

        float y = 0.0f;
#pragma unroll
        for (int n = 0; n < D_STATE; ++n) {
            h[n] = expf(dl * Arow[n]) * h[n] + du * bc[DT_RANK + n];
            y += h[n] * bc[DT_RANK + D_STATE + n];
        }
        y += u * Dd;
        const float z = xz[row * (long long)(2 * D_INNER) + D_INNER + d];
        y *= silu_f(z);
        y_h[row * D_INNER + d] = (half_t)y;
    }
}

// ---------------- host-side launch ----------------
static inline void* ws_alloc(char*& p, size_t bytes) {
    void* r = (void*)p;
    size_t aligned = (bytes + 255) & ~(size_t)255;
    p += aligned;
    return r;
}

extern "C" void kernel_launch(void* const* d_in, const int* in_sizes, int n_in,
                              void* d_out, int out_size, void* d_ws, size_t ws_size,
                              hipStream_t stream) {
    const float* x      = (const float*)d_in[0];   // [B,L,D_MODEL]
    const float* gamma  = (const float*)d_in[1];
    const float* beta   = (const float*)d_in[2];
    const float* W_in   = (const float*)d_in[3];   // [1024, 4096]
    const float* conv_w = (const float*)d_in[4];   // [2048, 4]
    const float* conv_b = (const float*)d_in[5];
    const float* W_x    = (const float*)d_in[6];   // [2048, 96]
    const float* W_dt   = (const float*)d_in[7];   // [64, 2048]
    const float* b_dt   = (const float*)d_in[8];
    const float* A_log  = (const float*)d_in[9];   // [2048, 16]
    const float* Dp     = (const float*)d_in[10];
    const float* W_out  = (const float*)d_in[11];  // [2048, 1024]
    float* out = (float*)d_out;                    // [B,L,D_MODEL]

    char* p = (char*)d_ws;
    half_t* xn_h    = (half_t*)ws_alloc(p, (size_t)NTOK * D_MODEL * 2);
    half_t* WinT_h  = (half_t*)ws_alloc(p, (size_t)(2 * D_INNER) * D_MODEL * 2);  // [4096][1024]
    float*  xz_f    = (float*) ws_alloc(p, (size_t)NTOK * 2 * D_INNER * 4);
    float*  xs_f    = (float*) ws_alloc(p, (size_t)NTOK * D_INNER * 4);
    half_t* xs_h    = (half_t*)ws_alloc(p, (size_t)NTOK * D_INNER * 2);
    half_t* WxT_h   = (half_t*)ws_alloc(p, (size_t)DBC_N * D_INNER * 2);          // [96][2048]
    float*  dbc_f   = (float*) ws_alloc(p, (size_t)NTOK * DBC_N * 4);
    half_t* dt_h    = (half_t*)ws_alloc(p, (size_t)NTOK * DT_RANK * 2);
    half_t* WdtT_h  = (half_t*)ws_alloc(p, (size_t)D_INNER * DT_RANK * 2);        // [2048][64]
    float*  draw_f  = (float*) ws_alloc(p, (size_t)NTOK * D_INNER * 4);
    float*  delta_f = (float*) ws_alloc(p, (size_t)NTOK * D_INNER * 4);
    half_t* y_h     = (half_t*)ws_alloc(p, (size_t)NTOK * D_INNER * 2);
    half_t* WoutT_h = (half_t*)ws_alloc(p, (size_t)D_MODEL * D_INNER * 2);        // [1024][2048]

    const int TPB = 256;
    auto cdiv = [](long long a, long long b) { return (int)((a + b - 1) / b); };

    // --- pack weights (transpose to [N][K] f16 so WMMA B-fragments are contiguous)
    k_transpose_f32_to_f16<<<cdiv((long long)D_MODEL * 2 * D_INNER, TPB), TPB, 0, stream>>>(
        W_in, WinT_h, /*R=*/D_MODEL, /*C=*/2 * D_INNER);
    k_transpose_f32_to_f16<<<cdiv((long long)D_INNER * DBC_N, TPB), TPB, 0, stream>>>(
        W_x, WxT_h, /*R=*/D_INNER, /*C=*/DBC_N);
    k_transpose_f32_to_f16<<<cdiv((long long)DT_RANK * D_INNER, TPB), TPB, 0, stream>>>(
        W_dt, WdtT_h, /*R=*/DT_RANK, /*C=*/D_INNER);
    k_transpose_f32_to_f16<<<cdiv((long long)D_INNER * D_MODEL, TPB), TPB, 0, stream>>>(
        W_out, WoutT_h, /*R=*/D_INNER, /*C=*/D_MODEL);

    // --- LayerNorm -> f16 activations
    k_layernorm<<<NTOK, 256, 0, stream>>>(x, gamma, beta, xn_h);

    dim3 gblk(32, 8);   // 8 waves, 64x64 block tile

    // --- GEMM1: xz = xn @ W_in   [2048 x 1024] x [1024 x 4096]
    {
        dim3 grid(NTOK / 64, cdiv(2 * D_INNER, 64));
        k_gemm_wmma<<<grid, gblk, 0, stream>>>(xn_h, WinT_h, xz_f, nullptr,
                                               NTOK, 2 * D_INNER, D_MODEL);
    }

    // --- depthwise causal conv + SiLU
    k_conv_silu<<<cdiv((long long)NTOK * D_INNER, TPB), TPB, 0, stream>>>(
        xz_f, conv_w, conv_b, xs_f, xs_h);

    // --- GEMM2: dbc = xs @ W_x   [2048 x 2048] x [2048 x 96]
    {
        dim3 grid(NTOK / 64, cdiv(DBC_N, 64));
        k_gemm_wmma<<<grid, gblk, 0, stream>>>(xs_h, WxT_h, dbc_f, nullptr,
                                               NTOK, DBC_N, D_INNER);
    }

    // --- dt slice -> f16
    k_extract_dt<<<cdiv((long long)NTOK * DT_RANK, TPB), TPB, 0, stream>>>(dbc_f, dt_h);

    // --- GEMM3: draw = dt @ W_dt   [2048 x 64] x [64 x 2048]
    {
        dim3 grid(NTOK / 64, cdiv(D_INNER, 64));
        k_gemm_wmma<<<grid, gblk, 0, stream>>>(dt_h, WdtT_h, draw_f, nullptr,
                                               NTOK, D_INNER, DT_RANK);
    }

    // --- delta = softplus(draw + b_dt)
    k_delta<<<cdiv((long long)NTOK * D_INNER, TPB), TPB, 0, stream>>>(draw_f, b_dt, delta_f);

    // --- selective scan + D skip + SiLU(z) gate
    k_scan<<<cdiv(B_SZ * D_INNER, TPB), TPB, 0, stream>>>(
        xs_f, delta_f, dbc_f, A_log, Dp, xz_f, y_h);

    // --- GEMM4: out = y @ W_out + residual   [2048 x 2048] x [2048 x 1024]
    {
        dim3 grid(NTOK / 64, cdiv(D_MODEL, 64));
        k_gemm_wmma<<<grid, gblk, 0, stream>>>(y_h, WoutT_h, out, x,
                                               NTOK, D_MODEL, D_INNER);
    }
}
